// ConvBERT_7825430413731
// MI455X (gfx1250) — matrix operand order repaired and checked by reading, not compiled, not run on hardware
//
#include <hip/hip_runtime.h>
#include <math.h>

// ---------------- problem constants ----------------
#define BATCH 4
#define SEQ   2048
#define DH    1536
#define NH    4          // effective attention heads
#define HS    192        // head size
#define AH    768        // all-head size
#define KW    7          // conv kernel
#define INTERD 3072
#define MTOK  (BATCH*SEQ)   // 8192 tokens

// ---------------- CDNA5 WMMA plumbing ----------------
typedef __attribute__((ext_vector_type(16))) __bf16 v16bf;
typedef __attribute__((ext_vector_type(8)))  float  v8f;
typedef __attribute__((ext_vector_type(8)))  unsigned short u16x8;

union BF16x16 { v16bf v; u16x8 h[2]; unsigned short u[16]; };

__device__ __forceinline__ unsigned short f2bf(float f) {
  unsigned int u = __float_as_uint(f);
  u += 0x7FFFu + ((u >> 16) & 1u);          // round-to-nearest-even
  return (unsigned short)(u >> 16);
}
__device__ __forceinline__ float bf2f(unsigned short h) {
  return __uint_as_float(((unsigned int)h) << 16);
}
__device__ __forceinline__ v8f wmma_bf16(v16bf a, v16bf b, v8f c) {
  return __builtin_amdgcn_wmma_f32_16x16x32_bf16(false, a, false, b, (short)0, c,
                                                 false, false);
}

// ---------------- conversion / packing kernels ----------------
__global__ void k_f2b(const float* __restrict__ s, unsigned short* __restrict__ d, long n) {
  for (long i = (long)blockIdx.x * blockDim.x + threadIdx.x; i < n;
       i += (long)gridDim.x * blockDim.x)
    d[i] = f2bf(s[i]);
}

// Pack an f32 weight into the WMMA B-fragment layout:
//   P[kb][nb][lane][e]  where  B[k][n]:  k = kb*32 + (lane>=16 ? 16:0) + e,
//                              n = nb*16 + (lane & 15)
// trans==0: W is (Kd x srcN) row-major;  trans==1: logical B[k][n] = W[n][k]
// (W is (srcN x Kd)).  Columns >= srcN are zero-filled (Wck 28 -> 32 pad).
__global__ void k_pack_w(const float* __restrict__ W, unsigned short* __restrict__ P,
                         int Kd, int N, int srcN, int trans) {
  long i = (long)blockIdx.x * blockDim.x + threadIdx.x;
  if (i >= (long)Kd * N) return;
  int e = (int)(i & 15);
  long g = i >> 4;
  int lane = (int)(g & 31); g >>= 5;
  int nb = (int)(g % (N >> 4));
  int kb = (int)(g / (N >> 4));
  int col = nb * 16 + (lane & 15);
  int krow = kb * 32 + ((lane >> 4) << 4) + e;
  float v = 0.f;
  if (col < srcN)
    v = trans ? W[(long)col * Kd + krow] : W[(long)krow * srcN + col];
  P[i] = f2bf(v);
}

// Pack V (bf16, [b*SEQ+s][h*HS+d]) into per-(b,h) B-fragment tiles:
//   P[(b,h)][kb=S/32][nb=12][lane][e] ; key = kb*32 + (lane>=16?16:0)+e,
//   dim = nb*16 + (lane&15)
__global__ void k_pack_v(const unsigned short* __restrict__ vsrc,
                         unsigned short* __restrict__ P) {
  long i = (long)blockIdx.x * blockDim.x + threadIdx.x;
  if (i >= (long)MTOK * AH) return;
  int e = (int)(i & 15);
  long g = i >> 4;
  int lane = (int)(g & 31); g >>= 5;
  int nb = (int)(g % 12); g /= 12;
  int kb = (int)(g % (SEQ / 32)); g /= (SEQ / 32);
  int h = (int)(g % NH);
  int b = (int)(g / NH);
  int key = kb * 32 + ((lane >> 4) << 4) + e;
  int dim = nb * 16 + (lane & 15);
  P[i] = vsrc[((long)b * SEQ + key) * AH + h * HS + dim];
}

__global__ void k_pad_bias(const float* __restrict__ b, float* __restrict__ bp) {
  int i = threadIdx.x;
  if (i < 32) bp[i] = (i < NH * KW) ? b[i] : 0.f;
}

// ---------------- depthwise conv (groups = Dh, pad = 3) ----------------
__global__ void k_dwconv(const float* __restrict__ x, const float* __restrict__ dw,
                         unsigned short* __restrict__ out) {
  long i = (long)blockIdx.x * blockDim.x + threadIdx.x;
  if (i >= (long)MTOK * DH) return;
  int c = (int)(i % DH);
  long t = i / DH;
  int s = (int)(t % SEQ);
  long b = t / SEQ;
  float acc = 0.f;
#pragma unroll
  for (int j = 0; j < KW; j++) {
    int ss = s + j - (KW / 2);
    if (ss >= 0 && ss < SEQ) acc += x[(b * SEQ + ss) * (long)DH + c] * dw[c * KW + j];
  }
  out[i] = f2bf(acc);
}

// ---------------- generic WMMA GEMM (software pipelined) ----------------
// C[M x N] = A[M x Kd](bf16 row-major) @ Wp(packed fragments) + bias
// Wave tile: 32 x (NT*16).  EPI: 0 none, 1 += resid, 2 exact GELU.
template <int NT, int EPI>
__global__ __launch_bounds__(256)
void k_gemm(const unsigned short* __restrict__ A, const unsigned short* __restrict__ Wp,
            const float* __restrict__ bias, const float* __restrict__ resid,
            float* __restrict__ outF, unsigned short* __restrict__ outB,
            int M, int N, int Kd) {
  int wave = threadIdx.x >> 5;
  int lane = threadIdx.x & 31;
  int m0 = (blockIdx.y * 8 + wave) * 32;
  int n0 = blockIdx.x * (NT * 16);
  const int NB16 = N >> 4;

  v8f acc[2][NT];
#pragma unroll
  for (int hh = 0; hh < 2; hh++)
#pragma unroll
    for (int j = 0; j < NT; j++)
#pragma unroll
      for (int i = 0; i < 8; i++) acc[hh][j][i] = 0.f;

  int rowA = lane & 15;
  int kbA  = (lane < 16) ? 0 : 16;                // A K-half per lane group
  const unsigned short* Ar0 = A + (long)(m0 + rowA) * Kd;
  const unsigned short* Ar1 = A + (long)(m0 + 16 + rowA) * Kd;
  const unsigned short* Bp  = Wp + (((long)(n0 >> 4) * 32 + lane) << 4);
  const long bstride = (long)NB16 * 512;          // elements per 32-K block

  BF16x16 a0[2], a1[2], bfrag[2][NT];             // double buffers
  auto ldA = [&](int k, int p) {
    a0[p].h[0] = *(const u16x8*)(Ar0 + k + kbA);
    a0[p].h[1] = *(const u16x8*)(Ar0 + k + 8 + kbA);
    a1[p].h[0] = *(const u16x8*)(Ar1 + k + kbA);
    a1[p].h[1] = *(const u16x8*)(Ar1 + k + 8 + kbA);
  };
  auto ldB = [&](int k, int p) {
    const unsigned short* bp = Bp + (long)(k >> 5) * bstride;
#pragma unroll
    for (int j = 0; j < NT; j++) {
      bfrag[p][j].h[0] = *(const u16x8*)(bp + (j << 9));
      bfrag[p][j].h[1] = *(const u16x8*)(bp + (j << 9) + 8);
    }
  };

  ldA(0, 0);
  ldB(0, 0);
  for (int k = 0; k < Kd; k += 32) {
    int p  = (k >> 5) & 1;
    int np = p ^ 1;
    if (k + 32 < Kd) {            // issue next tile's loads before this tile's math
      ldA(k + 32, np);
      ldB(k + 32, np);
      __builtin_prefetch(Bp + (long)((k >> 5) + 2) * bstride, 0, 1);
    }
#pragma unroll
    for (int j = 0; j < NT; j++) {
      acc[0][j] = wmma_bf16(a0[p].v, bfrag[p][j].v, acc[0][j]);
      acc[1][j] = wmma_bf16(a1[p].v, bfrag[p][j].v, acc[1][j]);
    }
  }

  int col = lane & 15;
  int rb  = (lane < 16) ? 0 : 8;
#pragma unroll
  for (int hh = 0; hh < 2; hh++)
#pragma unroll
    for (int j = 0; j < NT; j++) {
      int n = n0 + j * 16 + col;
      float bsv = bias ? bias[n] : 0.f;
#pragma unroll
      for (int i = 0; i < 8; i++) {
        int m = m0 + hh * 16 + rb + i;
        float v = acc[hh][j][i] + bsv;
        if (EPI == 1) v += resid[(long)m * N + n];
        if (EPI == 2) v = 0.5f * v * (1.0f + erff(v * 0.70710678f));
        if (outF) outF[(long)m * N + n] = v;
        if (outB) outB[(long)m * N + n] = f2bf(v);
      }
    }
}

// ---------------- elementwise conv_attn = key_conv * q ----------------
__global__ void k_mul(const unsigned short* __restrict__ a, const unsigned short* __restrict__ b,
                      unsigned short* __restrict__ o, long n) {
  long i = (long)blockIdx.x * blockDim.x + threadIdx.x;
  if (i < n) o[i] = f2bf(bf2f(a[i]) * bf2f(b[i]));
}

// ---------------- per-position kernel softmax (over K=7) ----------------
__global__ void k_ck_softmax(const float* __restrict__ raw, float* __restrict__ ck) {
  int i = blockIdx.x * blockDim.x + threadIdx.x;      // over MTOK * NH
  if (i >= MTOK * NH) return;
  int t = i >> 2, h = i & 3;
  const float* r = raw + (long)t * 32 + h * KW;       // raw padded to 32 cols
  float m = -3.0e38f;
#pragma unroll
  for (int k = 0; k < KW; k++) m = fmaxf(m, r[k]);
  float e[KW], s = 0.f;
#pragma unroll
  for (int k = 0; k < KW; k++) { e[k] = __expf(r[k] - m); s += e[k]; }
  float inv = 1.0f / s;
#pragma unroll
  for (int k = 0; k < KW; k++) ck[(long)t * (NH * KW) + h * KW + k] = e[k] * inv;
}

// ---------------- span-based dynamic convolution gather ----------------
__global__ void k_dynconv(const float* __restrict__ co, const float* __restrict__ ck,
                          float* __restrict__ ctx) {
  long i = (long)blockIdx.x * blockDim.x + threadIdx.x;   // over MTOK * AH
  if (i >= (long)MTOK * AH) return;
  int d = (int)(i % AH);
  long t = i / AH;
  int h = d / HS;
  int s = (int)(t % SEQ);
  long b = t / SEQ;
  const float* c = ck + t * (NH * KW) + h * KW;
  float acc = 0.f;
#pragma unroll
  for (int j = 0; j < KW; j++) {
    int ss = s + j - (KW / 2);
    if (ss >= 0 && ss < SEQ) acc += c[j] * co[(b * SEQ + ss) * (long)AH + d];
  }
  ctx[t * (long)DH + AH + d] = acc;
}

// ---------------- flash attention (WMMA, batched fragment loads) ----------------
// grid: (SEQ/64, NH, BATCH), block 128 (4 waves); wave w owns 16 queries.
// All 12 K-frags and 12 V-frags of a 32-key block are loaded up front so their
// latency overlaps the score WMMAs + softmax VALU + LDS bounce.
__global__ __launch_bounds__(128)
void k_attn(const unsigned short* __restrict__ Qm, const unsigned short* __restrict__ Km,
            const unsigned short* __restrict__ Vp, const float* __restrict__ amask,
            float* __restrict__ ctxF) {
  __shared__ float pLDS[4][16 * 32];
  int wv = threadIdx.x >> 5;
  int lane = threadIdx.x & 31;
  int qt = blockIdx.x * 4 + wv;
  int h = blockIdx.y;
  int b = blockIdx.z;
  long base = (long)b * SEQ * AH + h * HS;

  int rowA = lane & 15;
  int kbA  = (lane < 16) ? 0 : 16;
  int colB = lane & 15;
  int kbB  = (lane < 16) ? 0 : 16;
  int colC = lane & 15;
  int rb   = (lane < 16) ? 0 : 8;

  // Q fragments: 16 x 192 = 6 K-steps of 32 (vector loads, per-lane contiguous)
  BF16x16 qf[6];
  const unsigned short* qrow = Qm + base + (long)(qt * 16 + rowA) * AH;
#pragma unroll
  for (int f = 0; f < 6; f++) {
    qf[f].h[0] = *(const u16x8*)(qrow + f * 32 + kbA);
    qf[f].h[1] = *(const u16x8*)(qrow + f * 32 + 8 + kbA);
  }

  v8f acc[12];
#pragma unroll
  for (int n = 0; n < 12; n++)
#pragma unroll
    for (int i = 0; i < 8; i++) acc[n][i] = 0.f;
  float mrow[8], lrow[8];
#pragma unroll
  for (int i = 0; i < 8; i++) { mrow[i] = -3.0e38f; lrow[i] = 0.f; }

  const float scale = 0.07216878364f;  // 1/sqrt(192)
  const long vpbase = (((long)b * NH + h) * (SEQ / 32)) * (12 * 32 * 16);

  for (int kb0 = 0; kb0 < SEQ; kb0 += 32) {
    // ---- batch-load all K fragments of this 32-key block ----
    BF16x16 kf[2][6];
#pragma unroll
    for (int nb = 0; nb < 2; nb++) {
      const unsigned short* krow =
          Km + base + (long)(kb0 + nb * 16 + colB) * AH + kbB;
#pragma unroll
      for (int f = 0; f < 6; f++) {
        kf[nb][f].h[0] = *(const u16x8*)(krow + f * 32);
        kf[nb][f].h[1] = *(const u16x8*)(krow + f * 32 + 8);
      }
    }
    // ---- batch-load all V fragments (hidden under scores + softmax) ----
    BF16x16 vfr[12];
    const unsigned short* vt = Vp + vpbase + ((long)(kb0 >> 5) * (12 * 32 * 16));
#pragma unroll
    for (int n = 0; n < 12; n++) {
      const unsigned short* vp = vt + ((long)(n * 32 + lane) << 4);
      vfr[n].h[0] = *(const u16x8*)vp;
      vfr[n].h[1] = *(const u16x8*)(vp + 8);
    }
    if (kb0 + 32 < SEQ) {   // stream-ahead prefetch (global_prefetch_b8)
      __builtin_prefetch(Km + base + (long)(kb0 + 32 + colB) * AH, 0, 1);
      __builtin_prefetch(Km + base + (long)(kb0 + 48 + colB) * AH, 0, 1);
      __builtin_prefetch(vt + (12 * 32 * 16), 0, 1);
    }

    // ---- scores: 16 x 32 tile via 2 C-frags x 6 K-steps ----
    v8f sfr[2];
#pragma unroll
    for (int nb = 0; nb < 2; nb++) {
#pragma unroll
      for (int i = 0; i < 8; i++) sfr[nb][i] = 0.f;
#pragma unroll
      for (int f = 0; f < 6; f++)
        sfr[nb] = wmma_bf16(qf[f].v, kf[nb][f].v, sfr[nb]);
      float mb = (1.0f - amask[b * SEQ + kb0 + nb * 16 + colC]) * -1.0e30f;
#pragma unroll
      for (int i = 0; i < 8; i++) sfr[nb][i] = sfr[nb][i] * scale + mb;
    }
    // ---- online softmax (row stats via 16-lane butterflies) ----
    float mnew[8], corr[8];
#pragma unroll
    for (int i = 0; i < 8; i++) {
      float v = fmaxf(sfr[0][i], sfr[1][i]);
      for (int d = 1; d < 16; d <<= 1) v = fmaxf(v, __shfl_xor(v, d, 32));
      mnew[i] = fmaxf(mrow[i], v);
      corr[i] = __expf(mrow[i] - mnew[i]);
      mrow[i] = mnew[i];
    }
#pragma unroll
    for (int nb = 0; nb < 2; nb++)
#pragma unroll
      for (int i = 0; i < 8; i++) sfr[nb][i] = __expf(sfr[nb][i] - mnew[i]);
#pragma unroll
    for (int i = 0; i < 8; i++) {
      float s = sfr[0][i] + sfr[1][i];
      for (int d = 1; d < 16; d <<= 1) s += __shfl_xor(s, d, 32);
      lrow[i] = lrow[i] * corr[i] + s;
    }
#pragma unroll
    for (int n = 0; n < 12; n++)
#pragma unroll
      for (int i = 0; i < 8; i++) acc[n][i] *= corr[i];

    // ---- C-layout P -> A-layout bf16 via LDS ----
#pragma unroll
    for (int nb = 0; nb < 2; nb++)
#pragma unroll
      for (int i = 0; i < 8; i++) pLDS[wv][(rb + i) * 32 + nb * 16 + colC] = sfr[nb][i];
    __syncthreads();
    BF16x16 pf;
    {
      const float4* p0 = (const float4*)&pLDS[wv][rowA * 32 + kbA];
      const float4* p1 = (const float4*)&pLDS[wv][rowA * 32 + 8 + kbA];
      float4 r0 = p0[0], r1 = p0[1], r2 = p1[0], r3 = p1[1];
      pf.u[0] = f2bf(r0.x); pf.u[1] = f2bf(r0.y); pf.u[2] = f2bf(r0.z); pf.u[3] = f2bf(r0.w);
      pf.u[4] = f2bf(r1.x); pf.u[5] = f2bf(r1.y); pf.u[6] = f2bf(r1.z); pf.u[7] = f2bf(r1.w);
      pf.u[8]  = f2bf(r2.x); pf.u[9]  = f2bf(r2.y); pf.u[10] = f2bf(r2.z); pf.u[11] = f2bf(r2.w);
      pf.u[12] = f2bf(r3.x); pf.u[13] = f2bf(r3.y); pf.u[14] = f2bf(r3.z); pf.u[15] = f2bf(r3.w);
    }
    __syncthreads();

    // ---- P (16x32) @ V (32x192), V fragments already resident ----
#pragma unroll
    for (int n = 0; n < 12; n++)
      acc[n] = wmma_bf16(pf.v, vfr[n].v, acc[n]);
  }

  // ---- normalize + store into first half of ctx ----
#pragma unroll
  for (int n = 0; n < 12; n++)
#pragma unroll
    for (int i = 0; i < 8; i++) {
      long t = (long)b * SEQ + qt * 16 + rb + i;
      ctxF[t * DH + h * HS + n * 16 + colC] = acc[n][i] / lrow[i];
    }
}

// ---------------- LayerNorm (one block per row) ----------------
__global__ __launch_bounds__(256)
void k_ln(const float* __restrict__ x, const float* __restrict__ g,
          const float* __restrict__ bta, float* __restrict__ outF,
          unsigned short* __restrict__ outB) {
  __shared__ float sh[256];
  long row = blockIdx.x;
  const float* r = x + row * DH;
  float s = 0.f;
  for (int i = threadIdx.x; i < DH; i += 256) s += r[i];
  sh[threadIdx.x] = s; __syncthreads();
  for (int st = 128; st > 0; st >>= 1) {
    if ((int)threadIdx.x < st) sh[threadIdx.x] += sh[threadIdx.x + st];
    __syncthreads();
  }
  float mu = sh[0] * (1.0f / DH);
  __syncthreads();
  float v = 0.f;
  for (int i = threadIdx.x; i < DH; i += 256) { float d = r[i] - mu; v += d * d; }
  sh[threadIdx.x] = v; __syncthreads();
  for (int st = 128; st > 0; st >>= 1) {
    if ((int)threadIdx.x < st) sh[threadIdx.x] += sh[threadIdx.x + st];
    __syncthreads();
  }
  float rstd = rsqrtf(sh[0] * (1.0f / DH) + 1e-12f);
  for (int i = threadIdx.x; i < DH; i += 256) {
    float y = (r[i] - mu) * rstd * g[i] + bta[i];
    outF[row * DH + i] = y;
    if (outB) outB[row * DH + i] = f2bf(y);
  }
}

// ---------------- global max pool over sequence ----------------
__global__ void k_rowmax(const float* __restrict__ y, float* __restrict__ out) {
  int i = blockIdx.x * blockDim.x + threadIdx.x;   // over BATCH*DH
  if (i >= BATCH * DH) return;
  int d = i % DH, b = i / DH;
  float m = -3.0e38f;
  for (int s = 0; s < SEQ; s++) m = fmaxf(m, y[((long)b * SEQ + s) * DH + d]);
  out[i] = m;
}

// ==================== host launcher ====================
extern "C" void kernel_launch(void* const* d_in, const int* in_sizes, int n_in,
                              void* d_out, int out_size, void* d_ws, size_t ws_size,
                              hipStream_t stream) {
  (void)in_sizes; (void)n_in; (void)out_size; (void)ws_size;
  const float* x     = (const float*)d_in[0];
  const float* amask = (const float*)d_in[1];
  const float* Wq  = (const float*)d_in[2];  const float* bq  = (const float*)d_in[3];
  const float* Wk  = (const float*)d_in[4];  const float* bk  = (const float*)d_in[5];
  const float* Wv  = (const float*)d_in[6];  const float* bv  = (const float*)d_in[7];
  const float* dw  = (const float*)d_in[8];
  const float* pw  = (const float*)d_in[9];
  const float* sb  = (const float*)d_in[10];
  const float* Wck = (const float*)d_in[11]; const float* bck = (const float*)d_in[12];
  const float* Wco = (const float*)d_in[13]; const float* bco = (const float*)d_in[14];
  const float* Wo  = (const float*)d_in[15]; const float* bo  = (const float*)d_in[16];
  const float* g1  = (const float*)d_in[17]; const float* b1  = (const float*)d_in[18];
  const float* Wi  = (const float*)d_in[19]; const float* bi  = (const float*)d_in[20];
  const float* Wo2 = (const float*)d_in[21]; const float* bo2 = (const float*)d_in[22];
  const float* g2  = (const float*)d_in[23]; const float* b2  = (const float*)d_in[24];
  float* out = (float*)d_out;

  char* ws = (char*)d_ws;
  size_t off = 0;
  auto alloc = [&](size_t bytes) -> void* {
    void* p = (void*)(ws + off);
    off += (bytes + 255) & ~(size_t)255;
    return p;
  };
  typedef unsigned short u16;
  u16* xb    = (u16*)alloc((size_t)MTOK * DH * 2);
  u16* wq_p  = (u16*)alloc((size_t)DH * AH * 2);
  u16* wk_p  = (u16*)alloc((size_t)DH * AH * 2);
  u16* wv_p  = (u16*)alloc((size_t)DH * AH * 2);
  u16* pw_p  = (u16*)alloc((size_t)DH * AH * 2);
  u16* wck_p = (u16*)alloc((size_t)AH * 32 * 2);
  float* bck_p = (float*)alloc(32 * 4);
  u16* wco_p = (u16*)alloc((size_t)DH * AH * 2);
  u16* wo_p  = (u16*)alloc((size_t)DH * DH * 2);
  u16* wi_p  = (u16*)alloc((size_t)DH * INTERD * 2);
  u16* wo2_p = (u16*)alloc((size_t)INTERD * DH * 2);
  u16* qb    = (u16*)alloc((size_t)MTOK * AH * 2);
  u16* kb    = (u16*)alloc((size_t)MTOK * AH * 2);
  u16* vb    = (u16*)alloc((size_t)MTOK * AH * 2);
  u16* vpack = (u16*)alloc((size_t)MTOK * AH * 2);
  u16* dwo_b = (u16*)alloc((size_t)MTOK * DH * 2);
  u16* kc_b  = (u16*)alloc((size_t)MTOK * AH * 2);
  u16* ca_b  = (u16*)alloc((size_t)MTOK * AH * 2);
  float* ckraw = (float*)alloc((size_t)MTOK * 32 * 4);
  float* ck    = (float*)alloc((size_t)MTOK * NH * KW * 4);
  float* co_f  = (float*)alloc((size_t)MTOK * AH * 4);
  float* ctx_f = (float*)alloc((size_t)MTOK * DH * 4);
  u16*   ctx_b = (u16*)alloc((size_t)MTOK * DH * 2);
  float* y1_f  = (float*)alloc((size_t)MTOK * DH * 4);
  float* at_f  = (float*)alloc((size_t)MTOK * DH * 4);
  u16*   at_b  = (u16*)alloc((size_t)MTOK * DH * 2);
  u16*   int_b = (u16*)alloc((size_t)MTOK * INTERD * 2);
  float* y2_f  = (float*)alloc((size_t)MTOK * DH * 4);
  float* out_f = (float*)alloc((size_t)MTOK * DH * 4);

  auto cvt = [&](const float* s, u16* d, long n) {
    k_f2b<<<(int)((n + 255) / 256), 256, 0, stream>>>(s, d, n);
  };
  auto packw = [&](const float* W, u16* P, int Kd, int N, int srcN, int trans) {
    long n = (long)Kd * N;
    k_pack_w<<<(int)((n + 255) / 256), 256, 0, stream>>>(W, P, Kd, N, srcN, trans);
  };
  auto gemm = [&](const u16* A, const u16* Wp, const float* bias, const float* resid,
                  float* oF, u16* oB, int M, int N, int Kd, int epi) {
    dim3 g(N / 64, M / 256);
    if (epi == 0)
      k_gemm<4, 0><<<g, 256, 0, stream>>>(A, Wp, bias, resid, oF, oB, M, N, Kd);
    else if (epi == 1)
      k_gemm<4, 1><<<g, 256, 0, stream>>>(A, Wp, bias, resid, oF, oB, M, N, Kd);
    else
      k_gemm<4, 2><<<g, 256, 0, stream>>>(A, Wp, bias, resid, oF, oB, M, N, Kd);
  };

  // 1) precision conversion + fragment packing (weights resident in L2 after this)
  cvt(x, xb, (long)MTOK * DH);
  packw(Wq,  wq_p,  DH, AH, AH, 0);
  packw(Wk,  wk_p,  DH, AH, AH, 0);
  packw(Wv,  wv_p,  DH, AH, AH, 0);
  packw(pw,  pw_p,  DH, AH, AH, 1);            // pw^T
  packw(Wck, wck_p, AH, 32, NH * KW, 0);       // zero-padded 28 -> 32
  packw(Wco, wco_p, DH, AH, AH, 0);
  packw(Wo,  wo_p,  DH, DH, DH, 0);
  packw(Wi,  wi_p,  DH, INTERD, INTERD, 0);
  packw(Wo2, wo2_p, INTERD, DH, DH, 0);
  k_pad_bias<<<1, 32, 0, stream>>>(bck, bck_p);

  // 2) depthwise conv
  k_dwconv<<<((long)MTOK * DH + 255) / 256, 256, 0, stream>>>(x, dw, dwo_b);

  // 3) projections (WMMA GEMMs)
  gemm(xb,    wq_p, bq,  nullptr, nullptr, qb,   MTOK, AH, DH, 0);    // q
  gemm(xb,    wk_p, bk,  nullptr, nullptr, kb,   MTOK, AH, DH, 0);    // k
  gemm(xb,    wv_p, bv,  nullptr, nullptr, vb,   MTOK, AH, DH, 0);    // v
  gemm(dwo_b, pw_p, sb,  nullptr, nullptr, kc_b, MTOK, AH, DH, 0);    // key_conv
  gemm(xb,    wco_p, bco, nullptr, co_f,   nullptr, MTOK, AH, DH, 0); // co
  k_pack_v<<<((long)MTOK * AH + 255) / 256, 256, 0, stream>>>(vb, vpack);

  // 4) span-dynamic-conv branch
  k_mul<<<((long)MTOK * AH + 255) / 256, 256, 0, stream>>>(kc_b, qb, ca_b,
                                                           (long)MTOK * AH);
  {  // conv kernels (N = 32 padded)
    dim3 g(1, MTOK / 256);
    k_gemm<2, 0><<<g, 256, 0, stream>>>(ca_b, wck_p, bck_p, nullptr, ckraw, nullptr,
                                        MTOK, 32, AH);
  }
  k_ck_softmax<<<(MTOK * NH + 255) / 256, 256, 0, stream>>>(ckraw, ck);
  k_dynconv<<<((long)MTOK * AH + 255) / 256, 256, 0, stream>>>(co_f, ck, ctx_f);

  // 5) flash attention (WMMA) into first half of ctx
  {
    dim3 g(SEQ / 64, NH, BATCH);
    k_attn<<<g, 128, 0, stream>>>(qb, kb, vpack, amask, ctx_f);
  }

  // 6) output projection + residual + LN
  cvt(ctx_f, ctx_b, (long)MTOK * DH);
  gemm(ctx_b, wo_p, bo, x, y1_f, nullptr, MTOK, DH, DH, 1);
  k_ln<<<MTOK, 256, 0, stream>>>(y1_f, g1, b1, at_f, at_b);

  // 7) FFN (GELU epilogue) + residual + LN
  gemm(at_b, wi_p, bi, nullptr, nullptr, int_b, MTOK, INTERD, DH, 2);
  gemm(int_b, wo2_p, bo2, at_f, y2_f, nullptr, MTOK, DH, INTERD, 1);
  k_ln<<<MTOK, 256, 0, stream>>>(y2_f, g2, b2, out_f, nullptr);

  // 8) global max pool over sequence
  k_rowmax<<<(BATCH * DH + 255) / 256, 256, 0, stream>>>(out_f, out);
}
